// SEIMoE_82334523064731
// MI455X (gfx1250) — matrix operand order repaired and checked by reading, not compiled, not run on hardware
//
#include <hip/hip_runtime.h>
#include <hip/hip_bf16.h>

// ---------------------------------------------------------------------------
// MoE (top-2 of 8 experts + shared expert, sigmoid gate) for MI455X / gfx1250
// wave32, WMMA bf16 16x16x32 fp32-accum.
// Double-buffered software pipeline: async global->LDS (b128) + TDM tile
// loads for stage s+1 overlap WMMA compute on stage s.
// Weights/x pre-converted (weights pre-transposed to [N][K]) to bf16 once per
// call by bandwidth-bound kernels, so GEMM hot loops move raw bytes.
// ---------------------------------------------------------------------------

#define T_TOK   8192      // B*S
#define DIM_D   1024
#define DIM_F   2048
#define N_EXP   8
#define TOPK    2

#define BM      128       // token tile
#define BN      128       // output-col tile
#define BK      64        // K elements per pipeline stage (2 WMMA substeps)
#define XS      72        // LDS row stride (elements): 128B data + 16B pad
#define WS      72

typedef __attribute__((ext_vector_type(16))) __bf16 bf16x16;
typedef __attribute__((ext_vector_type(8)))  __bf16 bf16x8;
typedef __attribute__((ext_vector_type(8)))  float  floatx8;
typedef __attribute__((ext_vector_type(4)))  float  float4v;
typedef __attribute__((ext_vector_type(4)))  unsigned short ushort4v;
typedef __attribute__((ext_vector_type(4)))  unsigned int   uint4v;
typedef __attribute__((ext_vector_type(8)))  int            int8v;
typedef __attribute__((ext_vector_type(4)))  int            int4v;

__device__ __forceinline__ unsigned short f2bf_bits(float f) {
    unsigned u = __builtin_bit_cast(unsigned, f);
    u += 0x7FFFu + ((u >> 16) & 1u);            // round-to-nearest-even
    return (unsigned short)(u >> 16);
}

// Low 32 bits of a generic pointer to LDS == LDS byte offset (ISA 10.2).
__device__ __forceinline__ unsigned lds_addr32(const void* p) {
    return (unsigned)(unsigned long long)p;
}

// Async 16B copy global -> LDS (per-lane addresses, tracked by ASYNCcnt).
__device__ __forceinline__ void async_to_lds_b128(const void* g, void* l) {
    asm volatile("global_load_async_to_lds_b128 %0, %1, off"
                 :: "v"(lds_addr32(l)), "v"(g)
                 : "memory");
}
template <int N>
__device__ __forceinline__ void wait_asynccnt() {
    asm volatile("s_wait_asynccnt %0" :: "i"(N) : "memory");
}

// TDM: load a [dim1 x 64] bf16 2D tile (row stride `stride0` elements) into
// LDS, padding 4 DWORDs after every 32 DWORDs (-> 144B LDS row pitch = XS).
// tensor_dim clamps give hardware zero-fill for tail rows/cols.
__device__ __forceinline__ void tdm_load_tile_bf16(const void* gptr, void* lptr,
                                                   unsigned dim0_rem,
                                                   unsigned dim1_rem,
                                                   unsigned stride0) {
    unsigned long long ga = (unsigned long long)gptr;
    uint4v g0;
    g0[0] = 1u;                                   // count=1 valid descriptor
    g0[1] = lds_addr32(lptr);                     // lds_addr
    g0[2] = (unsigned)ga;                         // global_addr[31:0]
    g0[3] = (unsigned)((ga >> 32) & 0x1FFFFFFu)   // global_addr[56:32]
          | (2u << 30);                           // type=2 ("image")
    int8v g1;
    g1[0] = (int)((1u << 16)                      // data_size = 2 bytes
                | (1u << 20)                      // pad_enable
                | (4u << 22)                      // pad_interval: every 32 DW (128B)
                | (3u << 25));                    // pad_amount: 4 DW (16B)
    g1[1] = (int)((dim0_rem & 0xFFFFu) << 16);                    // tensor_dim0 lo
    g1[2] = (int)((dim0_rem >> 16) | ((dim1_rem & 0xFFFFu) << 16));
    g1[3] = (int)((dim1_rem >> 16) | ((unsigned)BK << 16));       // tile_dim0=64
    g1[4] = BM;                                                   // tile_dim1=128
    g1[5] = (int)stride0;                                         // dim0 stride lo
    g1[6] = 0;
    g1[7] = 0;
    int4v z4 = {0, 0, 0, 0};
#if __clang_major__ >= 23
    int8v z8 = {0, 0, 0, 0, 0, 0, 0, 0};
    __builtin_amdgcn_tensor_load_to_lds(g0, g1, z4, z4, z8, 0);
#else
    __builtin_amdgcn_tensor_load_to_lds(g0, g1, z4, z4, 0);
#endif
}

// Build a 16-element bf16 fragment from two 16-byte LDS chunks (ds_load_b128).
__device__ __forceinline__ bf16x16 make_frag(const __bf16* p0, const __bf16* p1) {
    bf16x8 lo = *reinterpret_cast<const bf16x8*>(p0);
    bf16x8 hi = *reinterpret_cast<const bf16x8*>(p1);
    bf16x16 r;
#pragma unroll
    for (int j = 0; j < 8; ++j) { r[j] = lo[j]; r[8 + j] = hi[j]; }
    return r;
}

// ---------------------------------------------------------------------------
// fp32 -> bf16 bulk convert (vectorized, grid-stride; n4 = n/4)
// ---------------------------------------------------------------------------
__global__ void convert_bf16_kernel(const float* __restrict__ src,
                                    unsigned short* __restrict__ dst,
                                    long long n4)
{
    long long i = (long long)blockIdx.x * blockDim.x + threadIdx.x;
    const long long stride = (long long)gridDim.x * blockDim.x;
    for (; i < n4; i += stride) {
        float4v v = ((const float4v*)src)[i];
        ushort4v o;
#pragma unroll
        for (int j = 0; j < 4; ++j) o[j] = f2bf_bits(v[j]);
        ((ushort4v*)dst)[i] = o;
    }
}

// ---------------------------------------------------------------------------
// fp32 [R,C] -> bf16 [C,R] tiled transpose-convert (batched over blockIdx.z)
// ---------------------------------------------------------------------------
__global__ void __launch_bounds__(256)
transpose_convert_kernel(const float* __restrict__ src,
                         unsigned short* __restrict__ dst, int R, int C)
{
    __shared__ unsigned short t[32][33];
    const size_t base = (size_t)blockIdx.z * R * C;
    const int c0 = blockIdx.x * 32, r0 = blockIdx.y * 32;
    const int tx = threadIdx.x & 31, ty = threadIdx.x >> 5;
#pragma unroll
    for (int j = 0; j < 4; ++j)
        t[ty + 8 * j][tx] = f2bf_bits(src[base + (size_t)(r0 + ty + 8 * j) * C + c0 + tx]);
    __syncthreads();
#pragma unroll
    for (int j = 0; j < 4; ++j)
        dst[base + (size_t)(c0 + ty + 8 * j) * R + r0 + tx] = t[tx][ty + 8 * j];
}

// ---------------------------------------------------------------------------
// Router: logits, softmax, top-2 atomic append, sigmoid gate. 1 wave / token.
// ---------------------------------------------------------------------------
__global__ void router_kernel(const float* __restrict__ x,
                              const float* __restrict__ rw,     // [E, D]
                              const float* __restrict__ gw,     // [1, D]
                              int*   __restrict__ counts,       // [E]
                              int*   __restrict__ tlist,        // [E, T]
                              float* __restrict__ clist,        // [E, T]
                              float* __restrict__ gsig)         // [T]
{
    const int tok  = (blockIdx.x * blockDim.x + threadIdx.x) >> 5;
    const int lane = threadIdx.x & 31;
    if (tok >= T_TOK) return;

    const float* xr = x + (size_t)tok * DIM_D;
    float acc[N_EXP + 1];
#pragma unroll
    for (int e = 0; e <= N_EXP; ++e) acc[e] = 0.f;

    for (int d = lane; d < DIM_D; d += 32) {
        const float xv = xr[d];
#pragma unroll
        for (int e = 0; e < N_EXP; ++e) acc[e] += xv * rw[e * DIM_D + d];
        acc[N_EXP] += xv * gw[d];
    }
#pragma unroll
    for (int off = 16; off > 0; off >>= 1) {
#pragma unroll
        for (int e = 0; e <= N_EXP; ++e) acc[e] += __shfl_xor(acc[e], off);
    }
    if (lane == 0) {
        float mx = acc[0];
#pragma unroll
        for (int e = 1; e < N_EXP; ++e) mx = fmaxf(mx, acc[e]);
        float p[N_EXP], sum = 0.f;
#pragma unroll
        for (int e = 0; e < N_EXP; ++e) { p[e] = __expf(acc[e] - mx); sum += p[e]; }
        const float inv = 1.f / sum;
#pragma unroll
        for (int e = 0; e < N_EXP; ++e) p[e] *= inv;

        int i0 = 0;
#pragma unroll
        for (int e = 1; e < N_EXP; ++e) if (p[e] > p[i0]) i0 = e;
        int i1 = (i0 == 0) ? 1 : 0;
#pragma unroll
        for (int e = 0; e < N_EXP; ++e) if (e != i0 && p[e] > p[i1]) i1 = e;

        int s0 = atomicAdd(&counts[i0], 1);
        tlist[i0 * T_TOK + s0] = tok;  clist[i0 * T_TOK + s0] = p[i0];
        int s1 = atomicAdd(&counts[i1], 1);
        tlist[i1 * T_TOK + s1] = tok;  clist[i1 * T_TOK + s1] = p[i1];

        gsig[tok] = 1.f / (1.f + __expf(-acc[N_EXP]));
    }
}

__global__ void offsets_kernel(const int* __restrict__ counts, int* __restrict__ offsets)
{
    if (threadIdx.x == 0) {
        int s = 0;
#pragma unroll
        for (int e = 0; e < N_EXP; ++e) { offsets[e] = s; s += counts[e]; }
    }
}

// ---------------------------------------------------------------------------
// gate/up GEMM + SiLU*up fusion, double-buffered async pipeline.
// A tile: gathered bf16 x rows; B tiles: pre-transposed bf16 weights [F][D].
// grid: (E or 1, T/128, F/128), block 256 (8 waves, 2x4 wave tiling).
// ---------------------------------------------------------------------------
__global__ void __launch_bounds__(256)
gateup_kernel(const unsigned short* __restrict__ xb,      // bf16 [T, D]
              const unsigned short* __restrict__ wgT_all, // bf16 [E?, F, D]
              const unsigned short* __restrict__ wuT_all,
              const int*   __restrict__ counts,
              const int*   __restrict__ offsets,
              const int*   __restrict__ tlist,
              unsigned short* __restrict__ hout,           // bf16 [rows, F]
              int shared_mode)
{
    __shared__ __align__(16) __bf16 sX [2][BM * XS];
    __shared__ __align__(16) __bf16 sWg[2][BN * WS];
    __shared__ __align__(16) __bf16 sWu[2][BN * WS];

    const int e = blockIdx.x;
    int count, hbase;
    const int* list;
    const unsigned short *wg, *wu;
    if (shared_mode) {
        count = T_TOK; hbase = 0; list = nullptr; wg = wgT_all; wu = wuT_all;
    } else {
        count = counts[e]; hbase = offsets[e]; list = tlist + e * T_TOK;
        wg = wgT_all + (size_t)e * DIM_D * DIM_F;
        wu = wuT_all + (size_t)e * DIM_D * DIM_F;
    }
    const int tile0 = blockIdx.y * BM;
    if (tile0 >= count) return;
    const int f0 = blockIdx.z * BN;

    const int tid  = threadIdx.x;
    const int wave = tid >> 5;
    const int lane = tid & 31;
    const int hf   = lane >> 4;
    const int lm   = lane & 15;
    const int wm   = wave >> 2;
    const int wn   = wave & 3;

    // Issue one pipeline stage (12 async instructions per wave).
    auto issue_stage = [&](int buf, int kk) {
#pragma unroll
        for (int j = 0; j < 4; ++j) {            // A: 128 rows x 8 chunks
            const int id = tid + 256 * j;        // 0..1023
            const int r = id >> 3, k8 = id & 7;
            int p = tile0 + r;
            if (p >= count) p = count - 1;       // clamp (tail rows discarded)
            const int t = list ? list[p] : p;
            async_to_lds_b128(xb + (size_t)t * DIM_D + kk + k8 * 8,
                              &sX[buf][r * XS + k8 * 8]);
        }
#pragma unroll
        for (int j = 0; j < 4; ++j) {            // B: 128 n-rows x 8 chunks
            const int id = tid + 256 * j;
            const int f = id >> 3, k8 = id & 7;
            const size_t g = (size_t)(f0 + f) * DIM_D + kk + k8 * 8;
            async_to_lds_b128(wg + g, &sWg[buf][f * WS + k8 * 8]);
            async_to_lds_b128(wu + g, &sWu[buf][f * WS + k8 * 8]);
        }
    };

    const floatx8 zero = {0.f,0.f,0.f,0.f,0.f,0.f,0.f,0.f};
    floatx8 accG[4][2], accU[4][2];
#pragma unroll
    for (int mi = 0; mi < 4; ++mi)
#pragma unroll
        for (int ni = 0; ni < 2; ++ni) { accG[mi][ni] = zero; accU[mi][ni] = zero; }

    constexpr int NSTAGE = DIM_D / BK;           // 16
    issue_stage(0, 0);
    int cur = 0;
    for (int s = 0; s < NSTAGE; ++s) {
        if (s + 1 < NSTAGE) {
            issue_stage(cur ^ 1, (s + 1) * BK);  // prefetch next stage
            wait_asynccnt<12>();                 // stage s landed; s+1 in flight
        } else {
            wait_asynccnt<0>();
        }
        __syncthreads();

#pragma unroll
        for (int ks = 0; ks < 2; ++ks) {         // two WMMA K-substeps
            const int ko = 32 * ks;
            bf16x16 aF[4];
#pragma unroll
            for (int mi = 0; mi < 4; ++mi) {
                const __bf16* b = &sX[cur][(wm * 64 + mi * 16 + lm) * XS + ko];
                aF[mi] = make_frag(b + 16 * hf, b + 16 * hf + 8);
            }
            bf16x16 bG[2], bU[2];
#pragma unroll
            for (int ni = 0; ni < 2; ++ni) {
                const __bf16* bg = &sWg[cur][(wn * 32 + ni * 16 + lm) * WS + ko];
                const __bf16* bu = &sWu[cur][(wn * 32 + ni * 16 + lm) * WS + ko];
                bG[ni] = make_frag(bg + 16 * hf, bg + 16 * hf + 8);
                bU[ni] = make_frag(bu + 16 * hf, bu + 16 * hf + 8);
            }
#pragma unroll
            for (int mi = 0; mi < 4; ++mi) {
#pragma unroll
                for (int ni = 0; ni < 2; ++ni) {
                    accG[mi][ni] = __builtin_amdgcn_wmma_f32_16x16x32_bf16(
                        false, aF[mi], false, bG[ni], (short)0, accG[mi][ni], false, false);
                    accU[mi][ni] = __builtin_amdgcn_wmma_f32_16x16x32_bf16(
                        false, aF[mi], false, bU[ni], (short)0, accU[mi][ni], false, false);
                }
            }
        }
        __syncthreads();                         // readers done -> buffer reusable
        cur ^= 1;
    }

#pragma unroll
    for (int mi = 0; mi < 4; ++mi) {
#pragma unroll
        for (int ni = 0; ni < 2; ++ni) {
#pragma unroll
            for (int r = 0; r < 8; ++r) {
                const int p = tile0 + wm * 64 + mi * 16 + hf * 8 + r;
                if (p < count) {
                    const int fc = f0 + wn * 32 + ni * 16 + lm;
                    const float g = accG[mi][ni][r];
                    const float u = accU[mi][ni][r];
                    const float h = (g / (1.f + __expf(-g))) * u;
                    hout[(size_t)(hbase + p) * DIM_F + fc] = f2bf_bits(h);
                }
            }
        }
    }
}

// ---------------------------------------------------------------------------
// down GEMM: A tile = dense h rows via TDM (zero-filled tails, LDS pitch via
// D# padding), B tile = transposed bf16 w_down via async b128; both double-
// buffered.  Scaled atomic scatter into output.
// grid: (E or 1, T/128, D/128), block 256.
// ---------------------------------------------------------------------------
__global__ void __launch_bounds__(256)
down_kernel(const unsigned short* __restrict__ h,        // bf16 [rows, F]
            const unsigned short* __restrict__ wdT_all,  // bf16 [E?, D, F]
            const int*   __restrict__ counts,
            const int*   __restrict__ offsets,
            const int*   __restrict__ tlist,
            const float* __restrict__ clist,             // [E, T]
            const float* __restrict__ gsig,              // [T]
            float* __restrict__ out,
            int shared_mode)
{
    __shared__ __align__(16) __bf16 sH[2][BM * XS];
    __shared__ __align__(16) __bf16 sW[2][BN * WS];

    const int e = blockIdx.x;
    int count, hbase;
    const int* list;
    const unsigned short* wd;
    const float* scale;
    if (shared_mode) {
        count = T_TOK; hbase = 0; list = nullptr; wd = wdT_all; scale = gsig;
    } else {
        count = counts[e]; hbase = offsets[e]; list = tlist + e * T_TOK;
        wd = wdT_all + (size_t)e * DIM_F * DIM_D;
        scale = clist + e * T_TOK;
    }
    const int tile0 = blockIdx.y * BM;
    if (tile0 >= count) return;
    const int d0 = blockIdx.z * BN;

    const int tid  = threadIdx.x;
    const int wave = tid >> 5;
    const int lane = tid & 31;
    const int hf   = lane >> 4;
    const int lm   = lane & 15;
    const int wm   = wave >> 2;
    const int wn   = wave & 3;

    // One pipeline stage: TDM for A (wave 0) + 4 async b128/wave for B.
    auto issue_stage = [&](int buf, int kk) {
        if (wave == 0) {
            tdm_load_tile_bf16(h + (size_t)(hbase + tile0) * DIM_F + kk, sH[buf],
                               (unsigned)(DIM_F - kk),
                               (unsigned)(count - tile0),
                               (unsigned)DIM_F);
        }
#pragma unroll
        for (int j = 0; j < 4; ++j) {
            const int id = tid + 256 * j;
            const int d = id >> 3, k8 = id & 7;
            async_to_lds_b128(wd + (size_t)(d0 + d) * DIM_F + kk + k8 * 8,
                              &sW[buf][d * WS + k8 * 8]);
        }
    };

    const floatx8 zero = {0.f,0.f,0.f,0.f,0.f,0.f,0.f,0.f};
    floatx8 acc[4][2];
#pragma unroll
    for (int mi = 0; mi < 4; ++mi)
#pragma unroll
        for (int ni = 0; ni < 2; ++ni) acc[mi][ni] = zero;

    constexpr int NSTAGE = DIM_F / BK;           // 32
    issue_stage(0, 0);
    int cur = 0;
    for (int s = 0; s < NSTAGE; ++s) {
        if (s + 1 < NSTAGE) {
            issue_stage(cur ^ 1, (s + 1) * BK);
            wait_asynccnt<4>();
            if (wave == 0) __builtin_amdgcn_s_wait_tensorcnt(1);
        } else {
            wait_asynccnt<0>();
            if (wave == 0) __builtin_amdgcn_s_wait_tensorcnt(0);
        }
        __syncthreads();

#pragma unroll
        for (int ks = 0; ks < 2; ++ks) {
            const int ko = 32 * ks;
            bf16x16 aF[4];
#pragma unroll
            for (int mi = 0; mi < 4; ++mi) {
                const __bf16* b = &sH[cur][(wm * 64 + mi * 16 + lm) * XS + ko];
                aF[mi] = make_frag(b + 16 * hf, b + 16 * hf + 8);
            }
            bf16x16 bF[2];
#pragma unroll
            for (int ni = 0; ni < 2; ++ni) {
                const __bf16* b = &sW[cur][(wn * 32 + ni * 16 + lm) * WS + ko];
                bF[ni] = make_frag(b + 16 * hf, b + 16 * hf + 8);
            }
#pragma unroll
            for (int mi = 0; mi < 4; ++mi)
#pragma unroll
                for (int ni = 0; ni < 2; ++ni)
                    acc[mi][ni] = __builtin_amdgcn_wmma_f32_16x16x32_bf16(
                        false, aF[mi], false, bF[ni], (short)0, acc[mi][ni], false, false);
        }
        __syncthreads();
        cur ^= 1;
    }

#pragma unroll
    for (int mi = 0; mi < 4; ++mi) {
#pragma unroll
        for (int ni = 0; ni < 2; ++ni) {
#pragma unroll
            for (int r = 0; r < 8; ++r) {
                const int p = tile0 + wm * 64 + mi * 16 + hf * 8 + r;
                if (p < count) {
                    const int t  = list ? list[p] : p;
                    const int dc = d0 + wn * 32 + ni * 16 + lm;
                    atomicAdd(&out[(size_t)t * DIM_D + dc], scale[p] * acc[mi][ni][r]);
                }
            }
        }
    }
}

// ---------------------------------------------------------------------------
// Host launcher
// ---------------------------------------------------------------------------
extern "C" void kernel_launch(void* const* d_in, const int* in_sizes, int n_in,
                              void* d_out, int out_size, void* d_ws, size_t ws_size,
                              hipStream_t stream)
{
    (void)in_sizes; (void)n_in; (void)ws_size;
    const float* x        = (const float*)d_in[0];
    const float* router_w = (const float*)d_in[1];
    const float* w_gate   = (const float*)d_in[2];
    const float* w_up     = (const float*)d_in[3];
    const float* w_down   = (const float*)d_in[4];
    const float* sw_gate  = (const float*)d_in[5];
    const float* sw_up    = (const float*)d_in[6];
    const float* sw_down  = (const float*)d_in[7];
    const float* gate_w   = (const float*)d_in[8];
    float* out = (float*)d_out;

    char* ws = (char*)d_ws;
    size_t off = 0;
    int*   counts  = (int*)  (ws + off); off += 256;
    int*   offsets = (int*)  (ws + off); off += 256;
    int*   tlist   = (int*)  (ws + off); off += (size_t)N_EXP * T_TOK * sizeof(int);
    float* clist   = (float*)(ws + off); off += (size_t)N_EXP * T_TOK * sizeof(float);
    float* gsig    = (float*)(ws + off); off += (size_t)T_TOK * sizeof(float);
    off = (off + 255) & ~(size_t)255;
    unsigned short* h_exp = (unsigned short*)(ws + off);
    off += (size_t)TOPK * T_TOK * DIM_F * sizeof(unsigned short);  // sum(counts)==2T
    unsigned short* h_sh  = (unsigned short*)(ws + off);
    off += (size_t)T_TOK * DIM_F * sizeof(unsigned short);
    unsigned short* xb    = (unsigned short*)(ws + off);
    off += (size_t)T_TOK * DIM_D * sizeof(unsigned short);
    unsigned short* wgT   = (unsigned short*)(ws + off);
    off += (size_t)N_EXP * DIM_D * DIM_F * sizeof(unsigned short);
    unsigned short* wuT   = (unsigned short*)(ws + off);
    off += (size_t)N_EXP * DIM_D * DIM_F * sizeof(unsigned short);
    unsigned short* wdT   = (unsigned short*)(ws + off);
    off += (size_t)N_EXP * DIM_F * DIM_D * sizeof(unsigned short);
    unsigned short* swgT  = (unsigned short*)(ws + off);
    off += (size_t)DIM_D * DIM_F * sizeof(unsigned short);
    unsigned short* swuT  = (unsigned short*)(ws + off);
    off += (size_t)DIM_D * DIM_F * sizeof(unsigned short);
    unsigned short* swdT  = (unsigned short*)(ws + off);
    off += (size_t)DIM_F * DIM_D * sizeof(unsigned short);

    hipMemsetAsync(counts, 0, 256, stream);
    hipMemsetAsync(out, 0, (size_t)out_size * sizeof(float), stream);

    // --- one-time-per-call precision/layout pass (bandwidth bound) ---
    convert_bf16_kernel<<<2048, 256, 0, stream>>>(x, xb, (long long)T_TOK * DIM_D / 4);
    transpose_convert_kernel<<<dim3(DIM_F/32, DIM_D/32, N_EXP), 256, 0, stream>>>(w_gate, wgT, DIM_D, DIM_F);
    transpose_convert_kernel<<<dim3(DIM_F/32, DIM_D/32, N_EXP), 256, 0, stream>>>(w_up,   wuT, DIM_D, DIM_F);
    transpose_convert_kernel<<<dim3(DIM_D/32, DIM_F/32, N_EXP), 256, 0, stream>>>(w_down, wdT, DIM_F, DIM_D);
    transpose_convert_kernel<<<dim3(DIM_F/32, DIM_D/32, 1), 256, 0, stream>>>(sw_gate, swgT, DIM_D, DIM_F);
    transpose_convert_kernel<<<dim3(DIM_F/32, DIM_D/32, 1), 256, 0, stream>>>(sw_up,   swuT, DIM_D, DIM_F);
    transpose_convert_kernel<<<dim3(DIM_D/32, DIM_F/32, 1), 256, 0, stream>>>(sw_down, swdT, DIM_F, DIM_D);

    // --- routing ---
    router_kernel<<<T_TOK / 8, 256, 0, stream>>>(x, router_w, gate_w,
                                                 counts, tlist, clist, gsig);
    offsets_kernel<<<1, 32, 0, stream>>>(counts, offsets);

    // --- expert + shared gate/up ---
    gateup_kernel<<<dim3(N_EXP, T_TOK / BM, DIM_F / BN), 256, 0, stream>>>(
        xb, wgT, wuT, counts, offsets, tlist, h_exp, 0);
    gateup_kernel<<<dim3(1, T_TOK / BM, DIM_F / BN), 256, 0, stream>>>(
        xb, swgT, swuT, counts, offsets, tlist, h_sh, 1);

    // --- down projections with scaled scatter-accumulate ---
    down_kernel<<<dim3(N_EXP, T_TOK / BM, DIM_D / BN), 256, 0, stream>>>(
        h_exp, wdT, counts, offsets, tlist, clist, gsig, out, 0);
    down_kernel<<<dim3(1, T_TOK / BM, DIM_D / BN), 256, 0, stream>>>(
        h_sh, swdT, counts, offsets, tlist, clist, gsig, out, 1);
}
